// FermiNet_57681410785852
// MI455X (gfx1250) — matrix operand order repaired and checked by reading, not compiled, not run on hardware
//
#include <hip/hip_runtime.h>
#include <math.h>

// ---------------------------------------------------------------------------
// FermiNet forward for MI455X (gfx1250, wave32, WMMA).
// One 256-thread workgroup (8 x wave32) per batch sample, grid = B = 1024.
// All matmuls with K,N >= 32 run on v_wmma_f32_16x16x32_f16 (f16 in, f32 acc):
//   - one-electron stream : (32 x K) @ (K x 256), K in {72,320,320,320}
//   - pair stream         : 4 blocks of (256 x h2) @ (h2 x 32) per layer
//   - orbital projection  : (32 x 256) @ (256 x 256)
//   - jastrow layer 0     : (32 x 256) @ (256 x 40->48)
// Weights are pre-converted/pre-swizzled into d_ws as f16 B-fragments
// ([kt][nt][lane][16], zero-padded K and N) by a prep kernel, so each
// B-fragment is a single aligned 32B load (2x global_load_b128) from L2.
// Activations live in LDS (~296KB of the 320KB WGP pool); A-fragments are
// 2x ds_load_b128. slogdet = per-lane partial-pivot LU of 16x16.
// ---------------------------------------------------------------------------

typedef __attribute__((ext_vector_type(16))) _Float16 v16h;
typedef __attribute__((ext_vector_type(8)))  float    v8f;

namespace fn {
constexpr int NE = 32, NUP = 16, SGL = 256, PR = 32, NDET = 16, NA = 4;
constexpr float INV_SQRT2 = 0.70710678118654752440f;
constexpr float SILU_GAIN = 1.7868f;

// LDS layout (byte offsets)
constexpr unsigned OFF_H1    = 0;       // 32x256 f32  131072
constexpr unsigned OFF_A16   = 131072;  // 32x320 f16   20480
constexpr unsigned OFF_H2A   = 151552;  // 32x32x32 f16 65536 (aliased: ORB f32 32x256)
constexpr unsigned OFF_H2B   = 217088;  // 32x32x32 f16 65536 (aliased: DETS f32 32x16x16)
constexpr unsigned OFF_PAIRM = 282624;  // 32x2x32 f32   8192 (aliased: J1 32x40 f32)
constexpr unsigned OFF_GUP   = 290816;  // 256 f32       1024 (aliased: cusp-diff reduce)
constexpr unsigned OFF_GDN   = 291840;  // 256 f32       1024
constexpr unsigned OFF_GNEW  = 292864;  // 2x256 f32     2048 (aliased: J2 32x6 f32)
constexpr unsigned OFF_RIM4  = 294912;  // 32x4x4 f32    2048
constexpr unsigned OFF_RIMN  = 296960;  // 32x4 f32       512
constexpr unsigned OFF_RIJN  = 297472;  // 32x32 f32     4096
constexpr unsigned OFF_ELE   = 301568;  // 32x3 f32       384
constexpr unsigned OFF_ATM   = 301952;  // 4x3 f32         64
constexpr unsigned OFF_REDA  = 302016;  // 32 f32         128
constexpr unsigned OFF_REDB  = 302144;  // 256 f32       1024
constexpr unsigned OFF_SLD   = 303168;  // 64 f32         256
constexpr unsigned OFF_MISC  = 303424;  // 16 f32          64
constexpr unsigned SMEM_BYTES = 303488; // ~296 KB / 320 KB WGP LDS

// workspace (f16 B-fragment) segment offsets, in halfs
constexpr int SEG_W1[4]   = {0, 24576, 106496, 188416};   // Kpad{96,320,320,320} x 256
constexpr int SEG_ORB     = 270336;                       // 256 x 256
constexpr int SEG_SAME[3] = {335872, 337920, 339968};     // Kpad 32 x 32
constexpr int SEG_DIFF[3] = {336896, 338944, 340992};
constexpr int SEG_J0      = 342016;                       // 256 x (40 -> 48)
constexpr int WS_HALFS    = 354304;                       // ~692 KB
} // namespace fn

struct P42 { const float* q[42]; };

__device__ __forceinline__ float silu_g(float x) {
  float s = 1.0f / (1.0f + __expf(-x));
  return x * s * fn::SILU_GAIN;
}
__device__ __forceinline__ float softplus_f(float x) {
  return (x > 20.0f) ? x : log1pf(__expf(x));
}
__device__ __forceinline__ v8f wmma16x16x32(v16h a, v16h b, v8f c) {
  return __builtin_amdgcn_wmma_f32_16x16x32_f16(false, a, false, b, (short)0, c,
                                                false, false);
}
// A-fragment (16x32 f16) K index for lane/element x:
//   elements 0..7 -> K 0..7 (+8 upper lane half), 8..15 -> K 16..23 (+8)
__device__ __forceinline__ int ka_of(int x, int aoff) {
  return ((x & 8) ? (x + 8) : x) + aoff;
}

// ---------------------------------------------------------------------------
// Weight prep: f32 [K x Ncols] row-major -> f16 fragments [kt][nt][lane][16],
// K zero-padded to a multiple of 32, Ncols zero-padded to NT*16.
// One output half per thread.
// ---------------------------------------------------------------------------
__global__ __launch_bounds__(256) void prep_weights(P42 P, _Float16* __restrict__ ws) {
  using namespace fn;
  int gid = blockIdx.x * 256 + threadIdx.x;
  if (gid >= WS_HALFS) return;
  const int off[13] = {0, 24576, 106496, 188416, 270336, 335872, 336896,
                       337920, 338944, 339968, 340992, 342016, 354304};
  const int qi[12]  = {4, 11, 18, 25, 28, 7, 9, 14, 16, 21, 23, 33};
  const int Ks[12]  = {72, 320, 320, 320, 256, 4, 4, 32, 32, 32, 32, 256};
  const int Nc[12]  = {256, 256, 256, 256, 256, 32, 32, 32, 32, 32, 32, 40};
  const int NT[12]  = {16, 16, 16, 16, 16, 2, 2, 2, 2, 2, 2, 3};
  int s = 0;
  while (gid >= off[s + 1]) ++s;
  int l    = gid - off[s];
  int x    = l & 15;
  int lane = (l >> 4) & 31;
  int ntk  = l >> 9;
  int nt   = ntk % NT[s];
  int kt   = ntk / NT[s];
  int k    = kt * 32 + ((x & 8) ? 16 : 0) + ((lane & 16) ? 8 : 0) + (x & 7);
  int col  = nt * 16 + (lane & 15);
  float v  = (k < Ks[s] && col < Nc[s]) ? P.q[qi[s]][k * Nc[s] + col] : 0.0f;
  ws[gid] = (_Float16)v;
}

__global__ __launch_bounds__(256) void ferminet_fwd(P42 P,
                                                    const _Float16* __restrict__ WS,
                                                    float* __restrict__ out) {
  using namespace fn;
  const int b    = blockIdx.x;
  const int tid  = threadIdx.x;
  const int lane = tid & 31;
  const int wave = tid >> 5;
  const int aoff = (lane & 16) ? 8 : 0;   // A-fragment K offset for upper lanes
  const int lnib = lane & 15;

  extern __shared__ char smem[];
  float*    H1    = (float*)   (smem + OFF_H1);
  _Float16* A16   = (_Float16*)(smem + OFF_A16);
  float*    PAIRM = (float*)(smem + OFF_PAIRM);
  float*    GUP   = (float*)(smem + OFF_GUP);
  float*    GDN   = (float*)(smem + OFF_GDN);
  float*    GNEW  = (float*)(smem + OFF_GNEW);
  float*    RIM4  = (float*)(smem + OFF_RIM4);
  float*    RIMN  = (float*)(smem + OFF_RIMN);
  float*    RIJN  = (float*)(smem + OFF_RIJN);
  float*    ELE   = (float*)(smem + OFF_ELE);
  float*    ATM   = (float*)(smem + OFF_ATM);
  float*    REDA  = (float*)(smem + OFF_REDA);
  float*    REDB  = (float*)(smem + OFF_REDB);
  float*    SLD   = (float*)(smem + OFF_SLD);
  float*    MISC  = (float*)(smem + OFF_MISC);

  // ---- load geometry ------------------------------------------------------
  const float* ele_g = P.q[0] + (size_t)b * (NE * 3);
  if (tid < NE * 3) ELE[tid] = ele_g[tid];
  if (tid < NA * 3) ATM[tid] = P.q[1][tid];
  __syncthreads();

  // ---- electron-nucleus features -----------------------------------------
  if (tid < NE * NA) {
    int e = tid >> 2, a = tid & 3;
    float dx = ELE[e * 3 + 0] - ATM[a * 3 + 0];
    float dy = ELE[e * 3 + 1] - ATM[a * 3 + 1];
    float dz = ELE[e * 3 + 2] - ATM[a * 3 + 2];
    float r  = sqrtf(dx * dx + dy * dy + dz * dz);
    float s  = log1pf(r) / r;
    RIMN[tid] = r;
    RIM4[tid * 4 + 0] = dx * s;
    RIM4[tid * 4 + 1] = dy * s;
    RIM4[tid * 4 + 2] = dz * s;
    RIM4[tid * 4 + 3] = r * s;
  }
  // ---- electron-electron features (h_two init, f16, padded to 32) --------
  for (int p = tid; p < NE * NE; p += 256) {
    int i = p >> 5, j = p & 31;
    float dx = ELE[i * 3 + 0] - ELE[j * 3 + 0];
    float dy = ELE[i * 3 + 1] - ELE[j * 3 + 1];
    float dz = ELE[i * 3 + 2] - ELE[j * 3 + 2];
    float ex = (i == j) ? 1.0f : 0.0f;
    float rx = dx + ex, ry = dy + ex, rz = dz + ex;
    float rn = sqrtf(rx * rx + ry * ry + rz * rz);
    float s  = log1pf(rn) / rn;
    float rc = (i == j) ? 0.0f : rn;
    RIJN[p] = rc;
    _Float16* h2p = (_Float16*)(smem + OFF_H2A) + p * PR;
    h2p[0] = (_Float16)(dx * s);
    h2p[1] = (_Float16)(dy * s);
    h2p[2] = (_Float16)(dz * s);
    h2p[3] = (_Float16)(rc * s);
    for (int f = 4; f < PR; ++f) h2p[f] = (_Float16)0.0f;
  }
  __syncthreads();

  // ---- nuclear embedding: h_one = mean_a tanh(rim4 @ emb[a] + bias[a]) ---
  for (int t = tid; t < NE * 64; t += 256) {
    int e = t >> 6, o = t & 63;
    float acc = 0.0f;
    for (int a = 0; a < NA; ++a) {
      const float* emb = P.q[2] + a * 256;  // [a][c][o], c<4, o<64
      float v = P.q[3][a * 64 + o];
      for (int c = 0; c < 4; ++c) v += RIM4[(e * 4 + a) * 4 + c] * emb[c * 64 + o];
      acc += tanhf(v);
    }
    H1[e * SGL + o] = 0.25f * acc;
  }
  __syncthreads();

  // ---- main layers --------------------------------------------------------
  const int IB1[4] = {5, 12, 19, 26}, IWG[4] = {6, 13, 20, 27};
  const int IBS[3] = {8, 15, 22},     IBD[3] = {10, 17, 24};
  int cur = 0;
  int h1dim = 64;
  for (int L = 0; L < 4; ++L) {
    const int h2in  = (L == 0) ? 4 : PR;
    const int fsh   = (L == 0) ? 2 : 5;
    const int Kin   = h1dim + 2 * h2in;        // 72 or 320
    const int Kpad  = (Kin + 31) & ~31;        // 96 or 320
    const int ktile = Kpad >> 5;
    // double-buffered h_two (select by byte offset, no LDS-pointer arrays)
    _Float16* h2c = (_Float16*)(smem + (cur ? OFF_H2B : OFF_H2A));
    _Float16* h2n = (_Float16*)(smem + (cur ? OFF_H2A : OFF_H2B));

    // warm L2/WGP$ for this layer's weight fragments (global_prefetch_b8)
    __builtin_prefetch(WS + SEG_W1[L] + (tid << 6), 0, 1);

    // (a) spin-resolved pair means (with up/down swap for e>=16)
    for (int t = tid; t < NE * h2in; t += 256) {
      int e = t >> fsh, f = t & (h2in - 1);
      float s0 = 0.0f, s1 = 0.0f;
      for (int j = 0; j < NUP; ++j) {
        s0 += (float)h2c[(e * NE + j) * PR + f];
        s1 += (float)h2c[(e * NE + NUP + j) * PR + f];
      }
      s0 *= (1.0f / 16.0f); s1 *= (1.0f / 16.0f);
      PAIRM[(e * 2 + 0) * PR + f] = (e < NUP) ? s0 : s1;
      PAIRM[(e * 2 + 1) * PR + f] = (e < NUP) ? s1 : s0;
    }
    // (b) global spin means of h_one
    if (tid < h1dim) {
      float su = 0.0f, sd = 0.0f;
      for (int e = 0; e < NUP; ++e) {
        su += H1[e * SGL + tid];
        sd += H1[(NUP + e) * SGL + tid];
      }
      GUP[tid] = su * (1.0f / 16.0f);
      GDN[tid] = sd * (1.0f / 16.0f);
    }
    __syncthreads();
    // (c) g_new = g_in @ w_g  (2 x 2h1 @ 2h1 x 256, scalar — tiny M)
    {
      const float* wg = P.q[IWG[L]];
      float g0 = 0.0f, g1 = 0.0f;
      for (int c = 0; c < h1dim; ++c) {
        float wu = wg[c * SGL + tid];
        float wl = wg[(h1dim + c) * SGL + tid];
        g0 += GUP[c] * wu + GDN[c] * wl;
        g1 += GDN[c] * wu + GUP[c] * wl;
      }
      GNEW[tid] = g0;
      GNEW[SGL + tid] = g1;
    }
    // (d) stage one_in = [h_one | pair0 | pair1 | 0-pad] as f16 WMMA A
    for (int e = 0; e < NE; ++e) {
      for (int k = tid; k < Kpad; k += 256) {
        float v;
        if (k < h1dim)             v = H1[e * SGL + k];
        else if (k < h1dim + h2in) v = PAIRM[(e * 2 + 0) * PR + (k - h1dim)];
        else if (k < Kin)          v = PAIRM[(e * 2 + 1) * PR + (k - h1dim - h2in)];
        else                       v = 0.0f;
        A16[e * Kpad + k] = (_Float16)v;
      }
    }
    __syncthreads();
    // (e,f) one-electron matmul: (32 x Kin) @ (Kin x 256) on WMMA, 32 tiles
    {
      const _Float16* wsb = WS + SEG_W1[L];
      const float* bone = P.q[IB1[L]];
      for (int qq = 0; qq < 4; ++qq) {
        int t  = wave + (qq << 3);
        int mt = t >> 4, nt = t & 15;
        v8f acc = {};
        int row = mt * 16 + lnib;
        for (int kt = 0; kt < ktile; ++kt) {
          v16h af;
#pragma unroll
          for (int x = 0; x < 16; ++x)
            af[x] = A16[row * Kpad + kt * 32 + ka_of(x, aoff)];
          v16h bf = *(const v16h*)(wsb + ((((kt << 4) + nt) << 5 | lane) << 4));
          acc = wmma16x16x32(af, bf, acc);
        }
        int n = (nt << 4) + lnib, mbase = mt * 16 + ((lane & 16) ? 8 : 0);
#pragma unroll
        for (int r = 0; r < 8; ++r) {
          int m = mbase + r;
          float v = acc[r] + bone[n] + GNEW[((m >= NUP) ? SGL : 0) + n];
          v = silu_g(v * INV_SQRT2);
          H1[m * SGL + n] = (L == 0) ? v : (H1[m * SGL + n] + v) * INV_SQRT2;
        }
      }
    }
    __syncthreads();
    // (g) pair update: 4 blocks of (256 x h2in) @ (h2in x 32) on WMMA
    if (L < 3) {
      const float* sb = P.q[IBS[L]];
      const float* db = P.q[IBD[L]];
      for (int qq = 0; qq < 16; ++qq) {
        int t   = wave + (qq << 3);        // 128 tiles total
        int blk = t >> 5, rem = t & 31;
        int mt  = rem >> 1, nt = rem & 1;
        bool same = (blk == 0) || (blk == 3);
        const _Float16* wsb = WS + (same ? SEG_SAME[L] : SEG_DIFF[L]);
        const float* bb = same ? sb : db;
        int ioff = (blk >= 2) ? NUP : 0;
        int joff = (blk & 1) ? NUP : 0;
        v8f acc = {};
        v16h af;
        {
          int i = ioff + mt, j = joff + lnib;  // A row = (i_loc*16 + j_loc)
          const _Float16* src = h2c + (i * NE + j) * PR;
#pragma unroll
          for (int x = 0; x < 16; ++x) af[x] = src[ka_of(x, aoff)];
        }
        v16h bf = *(const v16h*)(wsb + (((nt << 5) | lane) << 4));
        acc = wmma16x16x32(af, bf, acc);
        int n = (nt << 4) + lnib;
        int jbase = ((lane & 16) ? 8 : 0);
#pragma unroll
        for (int r = 0; r < 8; ++r) {
          int i = ioff + mt, j = joff + jbase + r;
          float v = acc[r] + bb[n];
          float o;
          if (L == 0) o = tanhf(v);
          else o = ((float)h2c[(i * NE + j) * PR + n] + silu_g(v)) * INV_SQRT2;
          h2n[(i * NE + j) * PR + n] = (_Float16)o;
        }
      }
      cur ^= 1;
      __syncthreads();
    }
    h1dim = SGL;
  }

  // ---- orbitals: phi = h_one @ orb_w + b, times envelope ------------------
  for (int t = tid; t < NE * SGL; t += 256) A16[t] = (_Float16)H1[t];
  __syncthreads();
  float* ORB = (float*)(smem + OFF_H2A);  // h_two buffers are dead now
  {
    const _Float16* wsb = WS + SEG_ORB;
    const float* ob  = P.q[29];
    const float* esg = P.q[30]; const float* epi = P.q[31];
    for (int qq = 0; qq < 4; ++qq) {
      int t  = wave + (qq << 3);
      int mt = t >> 4, nt = t & 15;
      v8f acc = {};
      int row = mt * 16 + lnib;
      for (int kt = 0; kt < 8; ++kt) {
        v16h af;
#pragma unroll
        for (int x = 0; x < 16; ++x)
          af[x] = A16[row * SGL + kt * 32 + ka_of(x, aoff)];
        v16h bf = *(const v16h*)(wsb + ((((kt << 4) + nt) << 5 | lane) << 4));
        acc = wmma16x16x32(af, bf, acc);
      }
      int n = (nt << 4) + lnib, mbase = mt * 16 + ((lane & 16) ? 8 : 0);
#pragma unroll
      for (int r = 0; r < 8; ++r) {
        int m = mbase + r;
        float phi = acc[r] + ob[n];
        float env = 0.0f;
        for (int a = 0; a < NA; ++a)
          env += __expf(-RIMN[m * 4 + a] * softplus_f(esg[a * SGL + n])) *
                 softplus_f(epi[a * SGL + n]);
        ORB[m * SGL + n] = phi * env;
      }
    }
  }
  __syncthreads();

  // ---- Jastrow layer 0 on WMMA: (32 x 256) @ (256 x 40->48) ---------------
  // A16 still holds final h_one as f16 fragments; 6 tiles on waves 0..5.
  float* J1 = PAIRM;  // 32x40 f32, PAIRM dead
  if (wave < 6) {
    const _Float16* wsb = WS + SEG_J0;
    const float* b0 = P.q[34];
    int mt = (wave >= 3) ? 1 : 0;
    int nt = wave - mt * 3;
    v8f acc = {};
    int row = mt * 16 + lnib;
    for (int kt = 0; kt < 8; ++kt) {
      v16h af;
#pragma unroll
      for (int x = 0; x < 16; ++x)
        af[x] = A16[row * SGL + kt * 32 + ka_of(x, aoff)];
      v16h bf = *(const v16h*)(wsb + (((kt * 3 + nt) << 5 | lane) << 4));
      acc = wmma16x16x32(af, bf, acc);
    }
    int n = (nt << 4) + lnib, mbase = mt * 16 + ((lane & 16) ? 8 : 0);
#pragma unroll
    for (int r = 0; r < 8; ++r) {
      int m = mbase + r;
      if (n < 40) J1[m * 40 + n] = silu_g(acc[r] + b0[n]);
    }
  }

  // ---- slogdet: one 16x16 partial-pivot LU per lane (32 dets) -------------
  float* DETS = (float*)(smem + OFF_H2B);
  if (tid < 2 * NDET) {
    int spin = tid >> 4, d = tid & 15;
    float* M = DETS + tid * 256;
    for (int i = 0; i < 16; ++i)
      for (int j = 0; j < 16; ++j)
        M[i * 16 + j] = ORB[(spin * NUP + i) * SGL + j * NDET + d];
    float sgn = 1.0f, ld = 0.0f;
    for (int k = 0; k < 16; ++k) {
      int p = k; float mx = fabsf(M[k * 16 + k]);
      for (int r = k + 1; r < 16; ++r) {
        float v = fabsf(M[r * 16 + k]);
        if (v > mx) { mx = v; p = r; }
      }
      if (p != k) {
        for (int c = k; c < 16; ++c) {
          float tv = M[k * 16 + c]; M[k * 16 + c] = M[p * 16 + c]; M[p * 16 + c] = tv;
        }
        sgn = -sgn;
      }
      float piv = M[k * 16 + k];
      ld += logf(fabsf(piv));
      if (piv < 0.0f) sgn = -sgn;
      float inv = 1.0f / piv;
      for (int r = k + 1; r < 16; ++r) {
        float f = M[r * 16 + k] * inv;
        for (int c = k + 1; c < 16; ++c) M[r * 16 + c] -= f * M[k * 16 + c];
      }
    }
    SLD[tid] = sgn;        // sign: up d -> [d], down d -> [16+d]
    SLD[32 + tid] = ld;    // logdet: [32+d] / [48+d]
  }
  __syncthreads();
  if (tid == 0) {
    const float* wd = P.q[32];
    float maxlog = -1e30f;
    for (int d = 0; d < NDET; ++d) {
      float l = SLD[32 + d] + SLD[48 + d];
      if (l > maxlog) maxlog = l;
    }
    float res = 0.0f;
    for (int d = 0; d < NDET; ++d) {
      float s = SLD[d] * SLD[16 + d];
      float l = SLD[32 + d] + SLD[48 + d];
      res += wd[d] * s * __expf(l - maxlog);
    }
    MISC[0] = logf(fabsf(res)) + maxlog;
  }

  // ---- Jastrow layers 1,2: 40 -> 6 -> 1 (scalar, tiny) --------------------
  float* J2 = GNEW;  // 192 f32, GNEW dead
  {
    const float* w1 = P.q[35]; const float* b1 = P.q[36];
    for (int t = tid; t < NE * 6; t += 256) {
      int e = t / 6, o = t - e * 6;
      float acc = b1[o];
      for (int c = 0; c < 40; ++c) acc += J1[e * 40 + c] * w1[c * 6 + o];
      J2[t] = silu_g(acc);
    }
  }
  __syncthreads();
  if (tid < NE) {
    const float* w2 = P.q[37];
    float acc = P.q[38][0];
    for (int c = 0; c < 6; ++c) acc += J2[tid * 6 + c] * w2[c];
    REDA[tid] = acc;
  }
  // ---- cusp sums ----------------------------------------------------------
  {
    float ap = P.q[41][0], aa = P.q[41][1];
    float ss = 0.0f, ds = 0.0f;
    for (int p = tid; p < NE * NE; p += 256) {
      int i = p >> 5, j = p & 31;
      float rn = RIJN[p];
      if ((i < NUP) == (j < NUP)) ss += ap * ap / (ap + rn);
      else                        ds += aa * aa / (aa + rn);
    }
    REDB[tid] = ss;
    GUP[tid]  = ds;  // GUP dead, reuse as second reduction buffer
  }
  __syncthreads();
  for (int s = 128; s > 0; s >>= 1) {
    if (tid < s) { REDB[tid] += REDB[tid + s]; GUP[tid] += GUP[tid + s]; }
    __syncthreads();
  }
  if (tid == 0) {
    float jm = 0.0f;
    for (int e = 0; e < NE; ++e) jm += REDA[e];
    jm *= (1.0f / 32.0f);
    float lp = MISC[0] + jm * P.q[39][0];
    lp -= 0.25f * P.q[40][0] * REDB[0] + 0.5f * P.q[40][1] * GUP[0];
    out[b] = lp;
  }
}

extern "C" void kernel_launch(void* const* d_in, const int* in_sizes, int n_in,
                              void* d_out, int out_size, void* d_ws, size_t ws_size,
                              hipStream_t stream) {
  (void)ws_size; (void)out_size;
  P42 P;
  for (int i = 0; i < 42; ++i) P.q[i] = (i < n_in) ? (const float*)d_in[i] : nullptr;
  int B = in_sizes[0] / (fn::NE * 3);
  _Float16* ws = (_Float16*)d_ws;
  int prep_blocks = (fn::WS_HALFS + 255) / 256;
  hipLaunchKernelGGL(prep_weights, dim3(prep_blocks), dim3(256), 0, stream, P, ws);
  (void)hipFuncSetAttribute(reinterpret_cast<const void*>(ferminet_fwd),
                            hipFuncAttributeMaxDynamicSharedMemorySize,
                            (int)fn::SMEM_BYTES);
  hipLaunchKernelGGL(ferminet_fwd, dim3(B), dim3(256), fn::SMEM_BYTES, stream,
                     P, (const _Float16*)ws, (float*)d_out);
}
